// DynamicQueryModule_64458869178914
// MI455X (gfx1250) — matrix operand order repaired and checked by reading, not compiled, not run on hardware
//
#include <hip/hip_runtime.h>
#include <hip/hip_bf16.h>

typedef _Float16 v16h __attribute__((ext_vector_type(16)));
typedef float    v8f  __attribute__((ext_vector_type(8)));

#define B_   8
#define C_   256
#define HW_  10000
#define HW4_ 2500
#define W_   100
#define H_   100

__device__ inline float clip5_(float v) { return fminf(5.f, fmaxf(-5.f, v)); }
__device__ inline float softplus_(float x) { return (x > 20.f) ? x : log1pf(expf(x)); }
__device__ inline float sigmoid_(float x) { return 1.f / (1.f + expf(-x)); }

// ---------------------------------------------------------------------------
// Kernel 1: per-(b,c) plane reductions over H*W with b128 loads.
//   df_avg / df_max of clipped density_feature, ef_gap of clipped encoder feat.
// ---------------------------------------------------------------------------
__global__ __launch_bounds__(256)
void reduce_bc_kernel(const float* __restrict__ df, const float* __restrict__ ef,
                      float* __restrict__ df_avg, float* __restrict__ df_max,
                      float* __restrict__ ef_gap)
{
    const int bc = blockIdx.x;                    // 0 .. B*C-1
    const float4* d4 = (const float4*)(df + (size_t)bc * HW_);
    const float4* e4 = (const float4*)(ef + (size_t)bc * HW_);
    float s = 0.f, mx = -1e30f, se = 0.f;
    for (int i = threadIdx.x; i < HW4_; i += 256) {
        float4 v = d4[i];
        float a0 = clip5_(v.x), a1 = clip5_(v.y), a2 = clip5_(v.z), a3 = clip5_(v.w);
        s += (a0 + a1) + (a2 + a3);
        mx = fmaxf(mx, fmaxf(fmaxf(a0, a1), fmaxf(a2, a3)));
        float4 e = e4[i];
        se += (clip5_(e.x) + clip5_(e.y)) + (clip5_(e.z) + clip5_(e.w));
    }
    __shared__ float rs[256], rm[256], re[256];
    rs[threadIdx.x] = s; rm[threadIdx.x] = mx; re[threadIdx.x] = se;
    __syncthreads();
    for (int off = 128; off > 0; off >>= 1) {
        if ((int)threadIdx.x < off) {
            rs[threadIdx.x] += rs[threadIdx.x + off];
            rm[threadIdx.x]  = fmaxf(rm[threadIdx.x], rm[threadIdx.x + off]);
            re[threadIdx.x] += re[threadIdx.x + off];
        }
        __syncthreads();
    }
    if (threadIdx.x == 0) {
        df_avg[bc] = rs[0] * (1.f / HW_);
        df_max[bc] = rm[0];
        ef_gap[bc] = re[0] * (1.f / HW_);
    }
}

// ---------------------------------------------------------------------------
// WMMA GEMM helper: Out[16][N] = relu(A[16][K] * W^T + bias)
//   A in LDS (f32, leading dim ldA), W global row-major [N][K].
//   One wave32; A and B both use contiguous k = 16*(lane/16) + e.
// ---------------------------------------------------------------------------
__device__ inline void wmma_gemm_relu(const float* __restrict__ Asrc, int ldA,
                                      const float* __restrict__ W,
                                      const float* __restrict__ bias,
                                      float* __restrict__ Out, int ldO,
                                      int K, int N, int lane)
{
    const int  m0 = lane & 15;
    const bool hi = lane >= 16;
    for (int nt = 0; nt < (N >> 4); ++nt) {
        v8f acc = {};
        for (int kt = 0; kt < (K >> 5); ++kt) {
            v16h a, b;
#pragma unroll
            for (int e = 0; e < 16; ++e) {
                int ka = (kt << 5) + e + (hi ? 16 : 0);
                a[e] = (_Float16)Asrc[m0 * ldA + ka];
                int kb = (kt << 5) + e + (hi ? 16 : 0);
                b[e] = (_Float16)W[(size_t)((nt << 4) + m0) * K + kb];
            }
            acc = __builtin_amdgcn_wmma_f32_16x16x32_f16(
                false, a, false, b, (short)0, acc, false, false);
        }
#pragma unroll
        for (int r = 0; r < 8; ++r) {
            int M = r + (hi ? 8 : 0);
            int Nc = (nt << 4) + m0;
            float v = acc[r] + bias[Nc];
            Out[M * ldO + Nc] = v > 0.f ? v : 0.f;
        }
    }
}

// ---------------------------------------------------------------------------
// Kernel 2: all per-batch heads in one wave. Uses V_WMMA_F32_16X16X32_F16.
// ---------------------------------------------------------------------------
__global__ __launch_bounds__(32)
void mlp_heads_kernel(const float* __restrict__ df_avg, const float* __restrict__ df_max,
                      const float* __restrict__ ef_gap,
                      const float* __restrict__ ln_in_g, const float* __restrict__ ln_in_b,
                      const float* __restrict__ fcb_w1,  const float* __restrict__ fcb_b1,
                      const float* __restrict__ fcb_w2,  const float* __restrict__ fcb_b2,
                      const float* __restrict__ cr_ln_g, const float* __restrict__ cr_ln_b,
                      const float* __restrict__ cr_w1,   const float* __restrict__ cr_b1,
                      const float* __restrict__ cr_w2,   const float* __restrict__ cr_b2,
                      const float* __restrict__ ca_w1,   const float* __restrict__ ca_b1,
                      const float* __restrict__ ca_w2,   const float* __restrict__ ca_b2,
                      float* __restrict__ out, float* __restrict__ ca_out)
{
    __shared__ float sA[16][512];       // 32 KB: LN'd gfeat, later LN'd feat_avg
    __shared__ float sH[16][256];       // 16 KB: hidden activations
    __shared__ float sRaw[B_][3];
    __shared__ float sBound[B_][3];
    __shared__ float sCa1[B_][16];
    const int lane = threadIdx.x;

    // ---- gfeat = [avg | max], rows >= 8 zero-padded ----
    for (int i = lane; i < 16 * 512; i += 32) {
        int r = i >> 9, c = i & 511;
        float v = 0.f;
        if (r < B_) v = (c < C_) ? df_avg[r * C_ + c] : df_max[r * C_ + (c - C_)];
        sA[r][c] = v;
    }
    __syncthreads();

    // ---- LayerNorm over 512 (rows 0..7) ----
    for (int r = 0; r < B_; ++r) {
        float s = 0.f, s2 = 0.f;
        for (int c = lane; c < 512; c += 32) { float v = sA[r][c]; s += v; s2 += v * v; }
        for (int off = 16; off > 0; off >>= 1) {
            s  += __shfl_down(s, off, 32);
            s2 += __shfl_down(s2, off, 32);
        }
        s = __shfl(s, 0, 32); s2 = __shfl(s2, 0, 32);
        float m = s * (1.f / 512.f);
        float inv = rsqrtf(s2 * (1.f / 512.f) - m * m + 1e-5f);
        for (int c = lane; c < 512; c += 32)
            sA[r][c] = (sA[r][c] - m) * inv * ln_in_g[c] + ln_in_b[c];
    }
    __syncthreads();

    // ---- fc_boundary layer 1: [16,512] x [512,256] -> relu ----
    wmma_gemm_relu(&sA[0][0], 512, fcb_w1, fcb_b1, &sH[0][0], 256, 512, 256, lane);
    __syncthreads();

    // ---- fc_boundary layer 2 (C->3), softplus*SCALE+5 cumsum ----
    if (lane < B_ * 3) {
        int b = lane / 3, j = lane % 3;
        float acc = fcb_b2[j];
        const float* w = fcb_w2 + j * C_;
        for (int c = 0; c < C_; ++c) acc += sH[b][c] * w[c];
        sRaw[b][j] = acc;
    }
    __syncthreads();
    if (lane < B_) {
        float cum = 0.f;
        for (int j = 0; j < 3; ++j) {
            float rv = sRaw[lane][j];
            cum += softplus_(rv) * 150.0f + 5.0f;      // SCALE = 1500/3*0.3
            sBound[lane][j] = cum;
            out[lane * 3 + j]      = cum;              // boundaries
            out[24 + lane * 3 + j] = rv;               // raw_boundaries
        }
    }
    __syncthreads();

    // ---- count regressor: LN(feat_avg) ----
    for (int i = lane; i < 16 * C_; i += 32) {
        int r = i >> 8, c = i & 255;
        sA[r][c] = (r < B_) ? df_avg[r * C_ + c] : 0.f;
    }
    __syncthreads();
    for (int r = 0; r < B_; ++r) {
        float s = 0.f, s2 = 0.f;
        for (int c = lane; c < C_; c += 32) { float v = sA[r][c]; s += v; s2 += v * v; }
        for (int off = 16; off > 0; off >>= 1) {
            s  += __shfl_down(s, off, 32);
            s2 += __shfl_down(s2, off, 32);
        }
        s = __shfl(s, 0, 32); s2 = __shfl(s2, 0, 32);
        float m = s * (1.f / 256.f);
        float inv = rsqrtf(s2 * (1.f / 256.f) - m * m + 1e-5f);
        for (int c = lane; c < C_; c += 32)
            sA[r][c] = (sA[r][c] - m) * inv * cr_ln_g[c] + cr_ln_b[c];
    }
    __syncthreads();

    // ---- count regressor layer 1: [16,256] x [256,256] -> relu ----
    wmma_gemm_relu(&sA[0][0], 512, cr_w1, cr_b1, &sH[0][0], 256, 256, 256, lane);
    __syncthreads();

    // ---- count regressor layer 2 + levels ----
    if (lane < B_) {
        float acc = cr_b2[0];
        for (int c = 0; c < C_; ++c) acc += sH[lane][c] * cr_w2[c];
        float pc = softplus_(acc);
        out[48 + lane] = pc;                            // pred_count
        int level = (pc > sBound[lane][0]) + (pc > sBound[lane][1]) + (pc > sBound[lane][2]);
        const float Q[4] = {300.f, 500.f, 900.f, 1500.f};
        out[56 + lane] = Q[level];                      // num_queries (as float)
    }

    // ---- CBAM channel attention: 256 -> 16 -> 256 ----
    for (int t = lane; t < B_ * 16; t += 32) {
        int b = t >> 4, o = t & 15;
        float acc = ca_b1[o];
        const float* w = ca_w1 + o * C_;
        const float* g = ef_gap + b * C_;
        for (int c = 0; c < C_; ++c) acc += g[c] * w[c];
        sCa1[b][o] = acc > 0.f ? acc : 0.f;
    }
    __syncthreads();
    for (int t = lane; t < B_ * C_; t += 32) {
        int b = t >> 8, c = t & 255;
        float acc = ca_b2[c];
        for (int o = 0; o < 16; ++o) acc += sCa1[b][o] * ca_w2[c * 16 + o];
        ca_out[b * C_ + c] = sigmoid_(acc);
    }
}

// ---------------------------------------------------------------------------
// Kernel 3: per-pixel channel stats of x = clip(ef)*ca :
//   mean_c, max_c (for CBAM spatial) and sum_c x*sw_w (fused 1x1 conv).
//   b128 loads, 4 pixels/thread, channels split over 4 thread-groups per
//   block (64 channels each), partials combined through LDS -> 4x the
//   memory-level parallelism of a flat channel loop.
// ---------------------------------------------------------------------------
__global__ __launch_bounds__(256)
void pixel_stats_kernel(const float* __restrict__ ef, const float* __restrict__ ca,
                        const float* __restrict__ sw_w,
                        float* __restrict__ sa_avg, float* __restrict__ sa_max,
                        float* __restrict__ dotp)
{
    __shared__ float sca[C_], ssw[C_];
    __shared__ float4 red_s[4][64], red_m[4][64], red_d[4][64];   // 12 KB
    const int tid  = threadIdx.x;
    const int b    = blockIdx.x / 40;
    const int qloc = tid & 63;
    const int grp  = tid >> 6;                       // 0..3 -> channel group
    const int q    = (blockIdx.x % 40) * 64 + qloc;  // float4 pixel-group index

    if (tid < C_) { sca[tid] = ca[b * C_ + tid]; ssw[tid] = sw_w[tid]; }
    __syncthreads();

    float4 s  = {0.f, 0.f, 0.f, 0.f};
    float4 mx = {-1e30f, -1e30f, -1e30f, -1e30f};
    float4 d  = {0.f, 0.f, 0.f, 0.f};
    if (q < HW4_) {
        const float4* base = (const float4*)(ef + (size_t)b * C_ * HW_) + q;
#pragma unroll 4
        for (int cc = 0; cc < 64; ++cc) {
            int c = (grp << 6) + cc;
            float4 v = base[(size_t)c * HW4_];
            float a = sca[c], w = ssw[c];
            float x0 = clip5_(v.x) * a, x1 = clip5_(v.y) * a;
            float x2 = clip5_(v.z) * a, x3 = clip5_(v.w) * a;
            s.x += x0; s.y += x1; s.z += x2; s.w += x3;
            mx.x = fmaxf(mx.x, x0); mx.y = fmaxf(mx.y, x1);
            mx.z = fmaxf(mx.z, x2); mx.w = fmaxf(mx.w, x3);
            d.x += x0 * w; d.y += x1 * w; d.z += x2 * w; d.w += x3 * w;
        }
    }
    red_s[grp][qloc] = s; red_m[grp][qloc] = mx; red_d[grp][qloc] = d;
    __syncthreads();

    if (grp == 0 && q < HW4_) {
#pragma unroll
        for (int g = 1; g < 4; ++g) {
            float4 ps = red_s[g][qloc], pm = red_m[g][qloc], pd = red_d[g][qloc];
            s.x += ps.x; s.y += ps.y; s.z += ps.z; s.w += ps.w;
            mx.x = fmaxf(mx.x, pm.x); mx.y = fmaxf(mx.y, pm.y);
            mx.z = fmaxf(mx.z, pm.z); mx.w = fmaxf(mx.w, pm.w);
            d.x += pd.x; d.y += pd.y; d.z += pd.z; d.w += pd.w;
        }
        float4 avg = {s.x * (1.f / C_), s.y * (1.f / C_), s.z * (1.f / C_), s.w * (1.f / C_)};
        ((float4*)(sa_avg + b * HW_))[q] = avg;
        ((float4*)(sa_max + b * HW_))[q] = mx;
        ((float4*)(dotp  + b * HW_))[q] = d;
    }
}

// ---------------------------------------------------------------------------
// Kernel 4: 7x7 spatial-attention conv (pad 3) + fused weight map:
//   wmap = sigmoid( sigmoid(conv) * dotp + sw_b )
// ---------------------------------------------------------------------------
__global__ __launch_bounds__(256)
void wmap_kernel(const float* __restrict__ sa_avg, const float* __restrict__ sa_max,
                 const float* __restrict__ dotp, const float* __restrict__ sa_w,
                 const float* __restrict__ sw_b, float* __restrict__ wmap)
{
    __shared__ float w0[49], w1[49];
    if (threadIdx.x < 49) {
        w0[threadIdx.x] = sa_w[threadIdx.x];
        w1[threadIdx.x] = sa_w[49 + threadIdx.x];
    }
    __syncthreads();
    const int b = blockIdx.x / 40;
    const int p = (blockIdx.x % 40) * 256 + threadIdx.x;
    if (p >= HW_) return;
    const int y = p / W_, x = p % W_;
    float acc = 0.f;
    for (int dy = 0; dy < 7; ++dy) {
        int yy = y + dy - 3;
        if (yy < 0 || yy >= H_) continue;
        for (int dx = 0; dx < 7; ++dx) {
            int xx = x + dx - 3;
            if (xx < 0 || xx >= W_) continue;
            int qq = b * HW_ + yy * W_ + xx;
            acc += sa_avg[qq] * w0[dy * 7 + dx] + sa_max[qq] * w1[dy * 7 + dx];
        }
    }
    float sig = sigmoid_(acc);
    wmap[b * HW_ + p] = sigmoid_(sig * dotp[b * HW_ + p] + sw_b[0]);
}

// ---------------------------------------------------------------------------
// Kernel 5: per-batch full bitonic sort (desc value, asc index tie-break,
//   exactly jax.lax.top_k ordering) of 10000 values padded to 16384 packed
//   u64 keys in 128 KB dynamic LDS, then emit top-1500 ref points.
// ---------------------------------------------------------------------------
__global__ __launch_bounds__(1024)
void topk_kernel(const float* __restrict__ wmap, float* __restrict__ ref_points)
{
    extern __shared__ unsigned long long keys[];   // 16384 * 8B = 128 KB
    const int NS = 16384;
    const int b = blockIdx.x;
    const int tid = threadIdx.x;

    for (int i = tid; i < NS; i += 1024) {
        if (i < HW_) {
            unsigned vb = __float_as_uint(wmap[b * HW_ + i]);   // sigmoid > 0: bits monotonic
            keys[i] = ((unsigned long long)vb << 32) | (unsigned)(0xFFFFFFFFu - (unsigned)i);
        } else {
            keys[i] = 0ull;
        }
    }
    __syncthreads();

    for (int k = 2; k <= NS; k <<= 1) {
        for (int j = k >> 1; j > 0; j >>= 1) {
            for (int i = tid; i < NS; i += 1024) {
                int ixj = i ^ j;
                if (ixj > i) {
                    unsigned long long a = keys[i], c = keys[ixj];
                    bool desc = ((i & k) == 0);
                    if ((a < c) == desc) { keys[i] = c; keys[ixj] = a; }
                }
            }
            __syncthreads();
        }
    }

    for (int p = tid; p < 1500; p += 1024) {
        unsigned idx = 0xFFFFFFFFu - (unsigned)(keys[p] & 0xFFFFFFFFull);
        float x = ((float)(idx % W_) + 0.5f) * (1.f / W_);
        float y = ((float)(idx / W_) + 0.5f) * (1.f / H_);
        float* o = ref_points + ((size_t)b * 1500 + p) * 4;
        o[0] = x; o[1] = y; o[2] = 0.05f; o[3] = 0.05f;
    }
}

// ---------------------------------------------------------------------------
extern "C" void kernel_launch(void* const* d_in, const int* in_sizes, int n_in,
                              void* d_out, int out_size, void* d_ws, size_t ws_size,
                              hipStream_t stream)
{
    const float* df      = (const float*)d_in[0];
    const float* ef      = (const float*)d_in[1];
    const float* ln_in_g = (const float*)d_in[2];
    const float* ln_in_b = (const float*)d_in[3];
    const float* fcb_w1  = (const float*)d_in[4];
    const float* fcb_b1  = (const float*)d_in[5];
    const float* fcb_w2  = (const float*)d_in[6];
    const float* fcb_b2  = (const float*)d_in[7];
    const float* cr_ln_g = (const float*)d_in[8];
    const float* cr_ln_b = (const float*)d_in[9];
    const float* cr_w1   = (const float*)d_in[10];
    const float* cr_b1   = (const float*)d_in[11];
    const float* cr_w2   = (const float*)d_in[12];
    const float* cr_b2   = (const float*)d_in[13];
    const float* ca_w1   = (const float*)d_in[14];
    const float* ca_b1   = (const float*)d_in[15];
    const float* ca_w2   = (const float*)d_in[16];
    const float* ca_b2   = (const float*)d_in[17];
    const float* sa_w    = (const float*)d_in[18];
    const float* sw_w    = (const float*)d_in[19];
    const float* sw_b    = (const float*)d_in[20];

    float* out = (float*)d_out;
    float* ws  = (float*)d_ws;
    float* df_avg = ws;                     // 2048
    float* df_max = ws + 2048;              // 2048
    float* ef_gap = ws + 4096;              // 2048
    float* ca     = ws + 6144;              // 2048
    float* sa_avg = ws + 8192;              // 80000
    float* sa_max = ws + 88192;             // 80000
    float* dotp   = ws + 168192;            // 80000

    // d_out layout: [0..24) boundaries, [24..48) raw, [48..56) pred_count,
    //               [56..64) num_queries, [64..80064) wmap, [80064..) ref_points
    float* o_wmap = out + 64;
    float* o_ref  = out + 64 + B_ * HW_;

    reduce_bc_kernel<<<B_ * C_, 256, 0, stream>>>(df, ef, df_avg, df_max, ef_gap);

    mlp_heads_kernel<<<1, 32, 0, stream>>>(df_avg, df_max, ef_gap,
                                           ln_in_g, ln_in_b,
                                           fcb_w1, fcb_b1, fcb_w2, fcb_b2,
                                           cr_ln_g, cr_ln_b, cr_w1, cr_b1, cr_w2, cr_b2,
                                           ca_w1, ca_b1, ca_w2, ca_b2,
                                           out, ca);

    pixel_stats_kernel<<<B_ * 40, 256, 0, stream>>>(ef, ca, sw_w, sa_avg, sa_max, dotp);

    wmap_kernel<<<B_ * 40, 256, 0, stream>>>(sa_avg, sa_max, dotp, sa_w, sw_b, o_wmap);

    topk_kernel<<<B_, 1024, 16384 * sizeof(unsigned long long), stream>>>(o_wmap, o_ref);
}
